// ElectrostaticAnalyzer_79534204387846
// MI455X (gfx1250) — matrix operand order repaired and checked by reading, not compiled, not run on hardware
//
#include <hip/hip_runtime.h>
#include <hip/hip_bf16.h>
#include <math.h>

#define B_ 4
#define L_ 384
#define F_ 640
#define H_ 128

typedef __attribute__((ext_vector_type(8)))  _Float16 h8;
typedef __attribute__((ext_vector_type(16))) _Float16 v16h;
typedef __attribute__((ext_vector_type(8)))  float    v8f;

// column of the one-hot charge table per amino acid (ACDEFGHIKLMNPQRSTVWY)
// charge>0 -> col0 (H,K,R), charge<0 -> col1 (D,E), else col2
__constant__ int kChgCol[20] = {2,2,1,1,2,2,0,2,0,2,2,2,2,2,0,2,2,2,2,2};

// ---------------------------------------------------------------------------
// k0: zero the energy accumulators (atomically accumulated by k3)
// ---------------------------------------------------------------------------
__global__ void k0_init(float* __restrict__ out_energy) {
    if (threadIdx.x < B_) out_energy[threadIdx.x] = 0.0f;
}

// ---------------------------------------------------------------------------
// k1: per-residue MLP (fp32 VALU; 0.75 GFLOP, L2-resident)
//   h1 = BN(relu(x@W1+b1)); h2 = relu(h1@W2+b2); pc = softmax(h2@W3+b3)
//   also fi = x@sW1[:F], fj = x@sW1[F:2F], pos/neg flags, net_charge (=qc)
// one block (128 threads) per residue row
// ---------------------------------------------------------------------------
__global__ __launch_bounds__(128)
void k1_residue_mlp(const float* __restrict__ features,
                    const int*   __restrict__ seq_ids,
                    const float* __restrict__ W1,  const float* __restrict__ b1,
                    const float* __restrict__ gamma, const float* __restrict__ beta,
                    const float* __restrict__ rmean, const float* __restrict__ rvar,
                    const float* __restrict__ W2,  const float* __restrict__ b2,
                    const float* __restrict__ W3,  const float* __restrict__ b3,
                    const float* __restrict__ sW1,
                    float* __restrict__ out_pc, float* __restrict__ out_tc,
                    float* __restrict__ out_nc,
                    float* __restrict__ ws_fi, float* __restrict__ ws_fj,
                    int*   __restrict__ ws_pos, int* __restrict__ ws_neg)
{
    __shared__ float xrow[F_];
    __shared__ float h1s[H_];
    __shared__ float h2s[64];

    const int row = blockIdx.x;            // 0 .. B*L-1
    const int tid = threadIdx.x;           // 0 .. 127
    const float* x = features + (size_t)row * F_;

    for (int k = tid; k < F_; k += 128) xrow[k] = x[k];
    __syncthreads();

    float a0 = 0.f, afi = 0.f, afj = 0.f;
    #pragma unroll 4
    for (int k = 0; k < F_; ++k) {
        const float xv = xrow[k];
        a0  = fmaf(xv, W1 [k*H_ + tid],        a0);
        afi = fmaf(xv, sW1[k*H_ + tid],        afi);
        afj = fmaf(xv, sW1[(F_ + k)*H_ + tid], afj);
    }
    float h1 = fmaxf(a0 + b1[tid], 0.f);
    h1 = gamma[tid] * (h1 - rmean[tid]) * rsqrtf(rvar[tid] + 1e-5f) + beta[tid];
    h1s[tid] = h1;
    ws_fi[(size_t)row * H_ + tid] = afi;
    ws_fj[(size_t)row * H_ + tid] = afj;
    __syncthreads();

    if (tid < 64) {
        float s = b2[tid];
        #pragma unroll 4
        for (int c = 0; c < H_; ++c) s = fmaf(h1s[c], W2[c*64 + tid], s);
        h2s[tid] = fmaxf(s, 0.f);
    }
    __syncthreads();

    if (tid == 0) {
        float lg[3];
        #pragma unroll
        for (int o = 0; o < 3; ++o) {
            float s = b3[o];
            for (int c = 0; c < 64; ++c) s = fmaf(h2s[c], W3[c*3 + o], s);
            lg[o] = s;
        }
        const float mx = fmaxf(lg[0], fmaxf(lg[1], lg[2]));
        const float e0 = expf(lg[0]-mx), e1 = expf(lg[1]-mx), e2 = expf(lg[2]-mx);
        const float inv = 1.f / (e0 + e1 + e2);
        const float p0 = e0*inv, p1 = e1*inv, p2 = e2*inv;
        out_pc[row*3+0] = p0; out_pc[row*3+1] = p1; out_pc[row*3+2] = p2;
        const int col = kChgCol[seq_ids[row]];
        out_tc[row*3+0] = (col==0) ? 1.f : 0.f;
        out_tc[row*3+1] = (col==1) ? 1.f : 0.f;
        out_tc[row*3+2] = (col==2) ? 1.f : 0.f;
        out_nc[row] = p0 - p1;                  // net_charge == qc
        ws_pos[row] = (p0 > p2) ? 1 : 0;
        ws_neg[row] = (p1 > p2) ? 1 : 0;
    }
}

// ---------------------------------------------------------------------------
// k2: pair MLP via WMMA (9.7 GFLOP, f16 in / f32 acc)
//   hid[m][c] = relu(fi[i][c] + fj[j0+m][c] + D*dvec[c] + sb1[c])  (16x128 f16)
//   hid2 = relu(hid @ sW2 + sb2)  -> 16 v_wmma_f32_16x16x32_f16 per tile
//   s = hid2 @ sW3 + sb3  folded into WMMA epilogue + shfl_xor reduce
//   q[i][j] = pos[i] & neg[j] & (D<4) & (sigmoid(s)>0.5)
// 8 waves / block, one 16-pair tile (fixed i, 16 consecutive j) per wave
// ---------------------------------------------------------------------------
__global__ __launch_bounds__(256)
void k2_pair_wmma(const float* __restrict__ structures,
                  const float* __restrict__ ws_fi, const float* __restrict__ ws_fj,
                  const float* __restrict__ sW1,   // dvec = row 2F
                  const float* __restrict__ sb1,
                  const float* __restrict__ sW2,  const float* __restrict__ sb2,
                  const float* __restrict__ sW3,  const float* __restrict__ sb3,
                  const int* __restrict__ ws_pos, const int* __restrict__ ws_neg,
                  unsigned char* __restrict__ ws_q)
{
    __shared__ _Float16 sW2T[64 * H_];       // [n][k], K-contiguous for B frags
    __shared__ _Float16 hidS[8][16 * H_];    // per-wave 16x128 A tile
    __shared__ float dls[H_], sb1s[H_], sb2s[64], sw3s[64];
    __shared__ float Dls[8][16];

    const int tid  = threadIdx.x;
    const int lane = tid & 31;
    const int wv   = tid >> 5;

    // stage sW2 transposed (f32 -> f16) + small constant vectors
    for (int idx = tid; idx < H_ * 64; idx += 256) {
        const int k = idx >> 6, n = idx & 63;
        sW2T[n*H_ + k] = (_Float16)sW2[k*64 + n];
    }
    if (tid < H_)            { dls[tid] = sW1[2*F_*H_ + tid]; sb1s[tid] = sb1[tid]; }
    else if (tid < H_ + 64)  { const int n = tid - H_; sb2s[n] = sb2[n]; sw3s[n] = sW3[n]; }
    __syncthreads();

    // tile id -> (b, i, j0)
    const int t    = blockIdx.x * 8 + wv;          // < 4*384*24 = 36864
    const int jt   = t % 24;
    const int i    = (t / 24) % L_;
    const int b    = t / (24 * L_);
    const int j0   = jt * 16;
    const int rowi = b * L_ + i;

    // distances for the 16 pairs of this tile
    {
        const int m = lane & 15;
        const float* si = structures + (size_t)rowi * 3;
        const float* sj = structures + (size_t)(b*L_ + j0 + m) * 3;
        const float dx = si[0]-sj[0], dy = si[1]-sj[1], dz = si[2]-sj[2];
        const float D = sqrtf(dx*dx + dy*dy + dz*dz + 1e-12f);
        if (lane < 16) Dls[wv][m] = D;
    }

    // build hid tile (fp32 math, f16 store)
    const float* fi = ws_fi + (size_t)rowi * H_;
    for (int idx = lane; idx < 16 * H_; idx += 32) {
        const int m = idx >> 7, c = idx & (H_ - 1);
        const float v = fi[c] + ws_fj[(size_t)(b*L_ + j0 + m)*H_ + c]
                        + Dls[wv][m] * dls[c] + sb1s[c];
        hidS[wv][idx] = (_Float16)fmaxf(v, 0.f);
    }

    // A fragments: 16-bit A 16x32 layout -> two contiguous 8-half groups/lane
    v16h afr[4];
    {
        const int m   = lane & 15;
        const int kb0 = 8 * (lane >> 4);
        const _Float16* hrow = &hidS[wv][m * H_];
        #pragma unroll
        for (int ks = 0; ks < 4; ++ks) {
            const int base = ks*32 + kb0;
            const h8 lo = *(const h8*)(hrow + base);
            const h8 hi = *(const h8*)(hrow + base + 16);
            afr[ks] = __builtin_shufflevector(lo, hi,
                        0,1,2,3,4,5,6,7,8,9,10,11,12,13,14,15);
        }
    }

    float partial[8];
    #pragma unroll
    for (int r = 0; r < 8; ++r) partial[r] = 0.f;

    const int nloc = lane & 15;
    const int kbB  = 8 * (lane >> 4);
    #pragma unroll
    for (int nt = 0; nt < 4; ++nt) {
        const int n = nloc + nt * 16;
        const _Float16* brow = &sW2T[n * H_];
        v8f acc = {0.f,0.f,0.f,0.f,0.f,0.f,0.f,0.f};
        #pragma unroll
        for (int ks = 0; ks < 4; ++ks) {
            const h8 lo = *(const h8*)(brow + ks*32 + kbB);
            const h8 hi = *(const h8*)(brow + ks*32 + kbB + 16);
            const v16h bfr = __builtin_shufflevector(lo, hi,
                        0,1,2,3,4,5,6,7,8,9,10,11,12,13,14,15);
            acc = __builtin_amdgcn_wmma_f32_16x16x32_f16(
                      false, afr[ks], false, bfr, (short)0, acc, false, false);
        }
        // epilogue: fold relu(+sb2) and the @sW3 dot product into the C layout
        // C/D f32 16x16: VGPR r -> pair m = r + 8*(lane>>4); N = (lane&15)+16*nt
        const float w3 = sw3s[n], bs = sb2s[n];
        #pragma unroll
        for (int r = 0; r < 8; ++r)
            partial[r] += fmaxf(acc[r] + bs, 0.f) * w3;
    }

    // reduce over the 16 N-lanes of each half-wave
    #pragma unroll
    for (int r = 0; r < 8; ++r) {
        float v = partial[r];
        v += __shfl_xor(v, 1, 32);
        v += __shfl_xor(v, 2, 32);
        v += __shfl_xor(v, 4, 32);
        v += __shfl_xor(v, 8, 32);
        partial[r] = v;
    }

    if ((lane & 15) == 0) {
        const float sb3v = sb3[0];
        const int posi   = ws_pos[rowi];
        const int mbase  = (lane >> 4) * 8;
        #pragma unroll
        for (int r = 0; r < 8; ++r) {
            const int m = mbase + r;
            const float s = partial[r] + sb3v;
            const float prob = 1.f / (1.f + expf(-s));
            const int j = j0 + m;
            const float D = Dls[wv][m];
            const int q = (posi && ws_neg[b*L_ + j] && (D < 4.f) && (prob > 0.5f)) ? 1 : 0;
            ws_q[((size_t)(b*L_ + i)) * L_ + j] = (unsigned char)q;
        }
    }
}

// ---------------------------------------------------------------------------
// k3: symmetrize mask + Coulomb energy (fp32 exact, block-reduced atomics)
// ---------------------------------------------------------------------------
__global__ __launch_bounds__(256)
void k3_mask_energy(const float* __restrict__ structures,
                    const float* __restrict__ nc,            // qc == net_charge
                    const unsigned char* __restrict__ ws_q,
                    float* __restrict__ out_mask,
                    float* __restrict__ out_energy)
{
    __shared__ float red[256];
    const int gid = blockIdx.x * 256 + threadIdx.x;   // < B*L*L, L*L % 256 == 0
    const int b   = gid / (L_ * L_);
    const int rem = gid % (L_ * L_);
    const int i   = rem / L_;
    const int j   = rem % L_;

    const unsigned char qij = ws_q[gid];
    const unsigned char qji = ws_q[(size_t)(b*L_ + j) * L_ + i];
    out_mask[gid] = (qij | qji) ? 1.f : 0.f;

    float e = 0.f;
    if (j > i) {
        const float* si = structures + (size_t)(b*L_ + i) * 3;
        const float* sj = structures + (size_t)(b*L_ + j) * 3;
        const float dx = si[0]-sj[0], dy = si[1]-sj[1], dz = si[2]-sj[2];
        const float D  = sqrtf(dx*dx + dy*dy + dz*dz + 1e-12f);
        if (D < 15.f && D > 0.f) {
            const float qi = nc[b*L_ + i], qj = nc[b*L_ + j];
            e = 332.f * qi * qj / ((80.f * 0.25f) * D * D);
        }
    }
    red[threadIdx.x] = e;
    __syncthreads();
    for (int s = 128; s > 0; s >>= 1) {
        if (threadIdx.x < s) red[threadIdx.x] += red[threadIdx.x + s];
        __syncthreads();
    }
    if (threadIdx.x == 0) atomicAdd(&out_energy[b], red[0]);
}

// ---------------------------------------------------------------------------
// k4: 1-D conv stack on net_charge (tiny). One block per batch, 384 threads.
// ---------------------------------------------------------------------------
__global__ __launch_bounds__(384)
void k4_conv(const float* __restrict__ nc,
             const float* __restrict__ c1w, const float* __restrict__ c1b,
             const float* __restrict__ c2w, const float* __restrict__ c2b,
             const float* __restrict__ c3w, const float* __restrict__ c3b,
             float* __restrict__ out_pot)
{
    __shared__ float ncs[L_];
    __shared__ float ch1[L_ * 16];
    __shared__ float ch2[L_ * 8];
    const int b = blockIdx.x;
    const int x = threadIdx.x;

    ncs[x] = nc[b*L_ + x];
    __syncthreads();

    #pragma unroll
    for (int o = 0; o < 16; ++o) {
        float s = c1b[o];
        #pragma unroll
        for (int t2 = 0; t2 < 5; ++t2) {
            const int xx = x + t2 - 2;
            if (xx >= 0 && xx < L_) s = fmaf(c1w[o*5 + t2], ncs[xx], s);
        }
        ch1[x*16 + o] = fmaxf(s, 0.f);
    }
    __syncthreads();

    #pragma unroll
    for (int o = 0; o < 8; ++o) {
        float s = c2b[o];
        #pragma unroll
        for (int t2 = 0; t2 < 3; ++t2) {
            const int xx = x + t2 - 1;
            if (xx >= 0 && xx < L_) {
                #pragma unroll
                for (int ic = 0; ic < 16; ++ic)
                    s = fmaf(c2w[o*48 + ic*3 + t2], ch1[xx*16 + ic], s);
            }
        }
        ch2[x*8 + o] = fmaxf(s, 0.f);
    }
    __syncthreads();

    float s = c3b[0];
    #pragma unroll
    for (int t2 = 0; t2 < 3; ++t2) {
        const int xx = x + t2 - 1;
        if (xx >= 0 && xx < L_) {
            #pragma unroll
            for (int ic = 0; ic < 8; ++ic)
                s = fmaf(c3w[ic*3 + t2], ch2[xx*8 + ic], s);
        }
    }
    out_pot[b*L_ + x] = s;
}

// ---------------------------------------------------------------------------
extern "C" void kernel_launch(void* const* d_in, const int* in_sizes, int n_in,
                              void* d_out, int out_size, void* d_ws, size_t ws_size,
                              hipStream_t stream) {
    const float* features   = (const float*)d_in[0];
    const int*   seq_ids    = (const int*)  d_in[1];
    const float* structures = (const float*)d_in[2];
    const float* W1   = (const float*)d_in[3];
    const float* b1   = (const float*)d_in[4];
    const float* gamma= (const float*)d_in[5];
    const float* beta = (const float*)d_in[6];
    const float* rmean= (const float*)d_in[7];
    const float* rvar = (const float*)d_in[8];
    const float* W2   = (const float*)d_in[9];
    const float* b2   = (const float*)d_in[10];
    const float* W3   = (const float*)d_in[11];
    const float* b3   = (const float*)d_in[12];
    const float* sW1  = (const float*)d_in[13];
    const float* sb1  = (const float*)d_in[14];
    const float* sW2  = (const float*)d_in[15];
    const float* sb2  = (const float*)d_in[16];
    const float* sW3  = (const float*)d_in[17];
    const float* sb3  = (const float*)d_in[18];
    const float* c1w  = (const float*)d_in[19];
    const float* c1b  = (const float*)d_in[20];
    const float* c2w  = (const float*)d_in[21];
    const float* c2b  = (const float*)d_in[22];
    const float* c3w  = (const float*)d_in[23];
    const float* c3b  = (const float*)d_in[24];

    // output layout (floats): pc | true_charges | net_charge | potential | mask | energy
    float* out = (float*)d_out;
    float* out_pc   = out;                    // 4608
    float* out_tc   = out + 4608;             // 4608
    float* out_nc   = out + 9216;             // 1536
    float* out_pot  = out + 10752;            // 1536
    float* out_mask = out + 12288;            // 589824
    float* out_en   = out + 602112;           // 4

    // workspace carve-up
    char* ws = (char*)d_ws;
    float* ws_fi = (float*)(ws + 0);                    // 1536*128 f32
    float* ws_fj = (float*)(ws + 786432);               // 1536*128 f32
    int*   ws_pos= (int*)  (ws + 1572864);              // 1536 i32
    int*   ws_neg= (int*)  (ws + 1579008);              // 1536 i32
    unsigned char* ws_q = (unsigned char*)(ws + 1585152); // 589824 u8

    k0_init<<<1, 32, 0, stream>>>(out_en);

    k1_residue_mlp<<<B_*L_, 128, 0, stream>>>(
        features, seq_ids, W1, b1, gamma, beta, rmean, rvar,
        W2, b2, W3, b3, sW1,
        out_pc, out_tc, out_nc, ws_fi, ws_fj, ws_pos, ws_neg);

    k2_pair_wmma<<<(B_*L_*24)/8, 256, 0, stream>>>(
        structures, ws_fi, ws_fj, sW1, sb1, sW2, sb2, sW3, sb3,
        ws_pos, ws_neg, ws_q);

    k3_mask_energy<<<(B_*L_*L_)/256, 256, 0, stream>>>(
        structures, out_nc, ws_q, out_mask, out_en);

    k4_conv<<<B_, L_, 0, stream>>>(
        out_nc, c1w, c1b, c2w, c2b, c3w, c3b, out_pot);
}
